// SpikingSynapticNeuralNetwork_29437705847084
// MI455X (gfx1250) — compile-verified, matrix operand order checked by reading, not run on hardware
//
#include <hip/hip_runtime.h>
#include <hip/hip_bf16.h>

typedef __attribute__((ext_vector_type(16))) _Float16 v16h;
typedef __attribute__((ext_vector_type(8)))  float    v8f;
typedef __attribute__((ext_vector_type(2)))  float    v2f;

#define B_TOT 16384
#define NI    256
#define NH    512
#define NO    128
#define NSTEP 25

#define ALPHA  0.9f
#define BETA   0.8f
#define THRESH 1.0f

union HFrag { v16h v; uint4 q[2]; unsigned int u[8]; _Float16 h[16]; };

// One block = 16 batch rows for all 25 timesteps.
// 8 waves (256 threads, wave32). Wave w owns output columns [w*16, w*16+16).
__global__ __launch_bounds__(256, 1)
void snn_fused_kernel(const float* __restrict__ x,
                      const float* __restrict__ W1,
                      const float* __restrict__ b1,
                      const float* __restrict__ W2,
                      const float* __restrict__ b2,
                      float* __restrict__ out)
{
    // LDS: 128K (W2 f16) + 32K (cur1) + 32K (syn1) + 32K (mem1) + 16K (spk1) = 240 KB
    __shared__ _Float16 ldsW2[NO * NH];     // [n][k], f16
    __shared__ float    lds_cur1[16 * NH];  // [m][h]
    __shared__ float    lds_syn1[16 * NH];
    __shared__ float    lds_mem1[16 * NH];
    __shared__ _Float16 lds_spk1[16 * NH];  // [m][k], f16 (A-matrix source)

    const int tid  = threadIdx.x;
    const int wave = tid >> 5;      // 0..7
    const int lane = tid & 31;
    const int l16  = lane & 15;
    const int hi   = lane >> 4;     // 0 or 1 (lane >= 16)
    const int row0 = blockIdx.x * 16;

    // ---- Phase 0: stage W2 (f32 -> f16) into LDS, zero layer-1 state ----
    for (int i = tid; i < NO * NH; i += 256)
        ldsW2[i] = (_Float16)W2[i];
    for (int i = tid; i < 16 * NH; i += 256) {
        lds_syn1[i] = 0.0f;
        lds_mem1[i] = 0.0f;
    }

    // ---- Phase 1: cur1 = x @ W1^T + b1 for this 16-row tile (exact f32 WMMA) ----
#if __has_builtin(__builtin_amdgcn_wmma_f32_16x16x4_f32)
    {
        // f32 WMMA 16x16x4. Wave w computes h-tiles {4w .. 4w+3} (NH/16 = 32 tiles).
        v8f acc[4];
        #pragma unroll
        for (int nt = 0; nt < 4; ++nt) acc[nt] = (v8f)(0.0f);

        const float* xrow = x + (size_t)(row0 + l16) * NI;   // A: lane = M
        #pragma unroll 4
        for (int kb = 0; kb < NI / 4; ++kb) {
            const int K = kb * 4 + hi * 2;                   // lanes>=16 hold K=2,3
            v2f a;
            a.x = xrow[K];
            a.y = xrow[K + 1];
            #pragma unroll
            for (int nt = 0; nt < 4; ++nt) {
                const int h = (wave * 4 + nt) * 16 + l16;    // B: lane = N
                const float* w1r = W1 + (size_t)h * NI;
                v2f b;
                b.x = w1r[K];
                b.y = w1r[K + 1];
                acc[nt] = __builtin_amdgcn_wmma_f32_16x16x4_f32(
                    false, a, false, b, (short)0, acc[nt], false, false);
            }
        }
        #pragma unroll
        for (int nt = 0; nt < 4; ++nt) {
            const int h = (wave * 4 + nt) * 16 + l16;
            const float bias = b1[h];
            #pragma unroll
            for (int r = 0; r < 8; ++r) {
                const int m = r + hi * 8;                    // C/D layout
                lds_cur1[m * NH + h] = acc[nt][r] + bias;
            }
        }
    }
#else
    {
        // Scalar fallback (one-time cost): each thread does 32 dot products.
        for (int i = 0; i < 32; ++i) {
            const int e = tid + i * 256;
            const int m = e >> 9, h = e & (NH - 1);
            const float* xr = x + (size_t)(row0 + m) * NI;
            const float* wr = W1 + (size_t)h * NI;
            float s = b1[h];
            for (int k = 0; k < NI; ++k) s = fmaf(xr[k], wr[k], s);
            lds_cur1[e] = s;
        }
    }
#endif
    __syncthreads();

    // ---- Phase 2: 25-step recurrence ----
    v8f syn2 = (v8f)(0.0f);
    v8f mem2 = (v8f)(0.0f);
    const int   n_out = wave * 16 + l16;
    const float bias2 = b2[n_out];

    const size_t SYN_OFF = (size_t)NSTEP * B_TOT * NO;
    const size_t MEM_OFF = 2 * SYN_OFF;

    for (int t = 0; t < NSTEP; ++t) {
        // Layer-1 synaptic neuron update (elementwise over [16, NH] in LDS)
        #pragma unroll 8
        for (int i = 0; i < 32; ++i) {
            const int e = tid + i * 256;
            float s = lds_syn1[e];
            float m = lds_mem1[e];
            const float c = lds_cur1[e];
            const float reset = (m > THRESH) ? THRESH : 0.0f;
            s = ALPHA * s + c;
            m = BETA * m + s - reset;
            lds_syn1[e] = s;
            lds_mem1[e] = m;
            lds_spk1[e] = (_Float16)((m > THRESH) ? 1.0f : 0.0f);
        }
        __syncthreads();

        // cur2 = spk1 @ W2^T via V_WMMA_F32_16X16X32_F16 (binary spk1 exact in f16)
        v8f c2 = (v8f)(0.0f);
        const _Float16* arow = lds_spk1 + l16 * NH;            // A: lane = M
        const _Float16* brow = ldsW2 + (size_t)n_out * NH;     // B: lane = N
        #pragma unroll 4
        for (int kb = 0; kb < NH / 32; ++kb) {
            const int K0 = kb * 32 + hi * 8;                   // ISA 16-bit A/B layout
            HFrag a, b;
            // two 16B-aligned contiguous chunks each -> ds_load_b128
            a.q[0] = *(const uint4*)(arow + K0);
            a.q[1] = *(const uint4*)(arow + K0 + 16);
            b.q[0] = *(const uint4*)(brow + K0);
            b.q[1] = *(const uint4*)(brow + K0 + 16);
            c2 = __builtin_amdgcn_wmma_f32_16x16x32_f16(
                false, a.v, false, b.v, (short)0, c2, false, false);
        }

        // Layer-2 neuron update in C-fragment registers + record outputs.
        // Output stream is write-once (629 MB total): store non-temporally so it
        // doesn't evict x/W1/W2 from L2.
        #pragma unroll
        for (int r = 0; r < 8; ++r) {
            const float cur = c2[r] + bias2;
            float sv = syn2[r];
            float mv = mem2[r];
            const float reset = (mv > THRESH) ? THRESH : 0.0f;
            sv = ALPHA * sv + cur;
            mv = BETA * mv + sv - reset;
            const float spike = (mv > THRESH) ? 1.0f : 0.0f;
            syn2[r] = sv;
            mem2[r] = mv;
            const size_t idx = (size_t)t * B_TOT * NO
                             + (size_t)(row0 + r + hi * 8) * NO + n_out;
            __builtin_nontemporal_store(spike, &out[idx]);
            __builtin_nontemporal_store(sv,    &out[SYN_OFF + idx]);
            __builtin_nontemporal_store(mv,    &out[MEM_OFF + idx]);
        }
        __syncthreads();   // protect lds_spk1 before next step rewrites it
    }
}

extern "C" void kernel_launch(void* const* d_in, const int* in_sizes, int n_in,
                              void* d_out, int out_size, void* d_ws, size_t ws_size,
                              hipStream_t stream) {
    const float* x  = (const float*)d_in[0];
    const float* W1 = (const float*)d_in[1];
    const float* b1 = (const float*)d_in[2];
    const float* W2 = (const float*)d_in[3];
    const float* b2 = (const float*)d_in[4];
    float* out = (float*)d_out;

    dim3 grid(B_TOT / 16);
    dim3 block(256);
    snn_fused_kernel<<<grid, block, 0, stream>>>(x, W1, b1, W2, b2, out);
}
